// net_prot_gat_17806934409266
// MI455X (gfx1250) — compile-verified, matrix operand order checked by reading, not run on hardware
//
#include <hip/hip_runtime.h>
#include <hip/hip_bf16.h>

typedef __bf16 bf16;
typedef __attribute__((ext_vector_type(16))) __bf16 v16bf;
typedef __attribute__((ext_vector_type(8)))  __bf16 v8bf;
typedef __attribute__((ext_vector_type(8)))  float  v8f;

#define N_NODES 1000
#define NP      1008            // padded N (63 tiles of 16)
#define NPX     1024            // j-stride of transposed x (contains all frag reads)
#define DIM     256
#define NL      7
#define NB      64
#define MP      (NB*NP)         // 64512 padded rows
#define DD      (DIM*DIM)

__device__ __forceinline__ bf16 f2bf(float f) {
  union { float f; unsigned u; } v; v.f = f;
  unsigned r = v.u + 0x7FFFu + ((v.u >> 16) & 1u);   // round-to-nearest-even
  unsigned short h = (unsigned short)(r >> 16);
  return __builtin_bit_cast(bf16, h);
}

// Load a 16-element bf16 WMMA fragment for this lane.
// base = row_ptr + k0 + 8*half ; runs are [0..7] and [16..23] from base.
__device__ __forceinline__ v16bf load_frag(const bf16* base) {
  v8bf lo = *(const v8bf*)(base);
  v8bf hi = *(const v8bf*)(base + 16);
  return __builtin_shufflevector(lo, hi, 0,1,2,3,4,5,6,7,8,9,10,11,12,13,14,15);
}

#define WMMA_BF16(a, b, c) \
  __builtin_amdgcn_wmma_f32_16x16x32_bf16(false, (a), false, (b), (short)0, (c), false, false)

// ---------------------------------------------------------------------------
// GEMM: out = act(A[MP,DIM](bf16) @ W^T + bias) (+res). W is [out,in] bf16.
// Block = 256 thr (8 waves), tile 128x64; wave = 32 rows x 32 cols
// (2x2 WMMA tiles -> 4 WMMAs per 4 fragment loads).
// OUTXT also writes the transposed bf16 copy XT[b][col][n] (stride NPX).
// ---------------------------------------------------------------------------
template<int OUTF, int OUTBF, int OUTXT, int RELU, int RES, int TO_DOUT>
__global__ __launch_bounds__(256)
void gemm_kernel(const bf16* __restrict__ A, const bf16* __restrict__ W,
                 const float* __restrict__ bias, const float* __restrict__ res,
                 float* __restrict__ outf, bf16* __restrict__ outbf,
                 bf16* __restrict__ xt)
{
  const int lane = threadIdx.x & 31;
  const int wave = threadIdx.x >> 5;
  const int lp = lane & 15, half = lane >> 4;
  const int wy = wave >> 1, wx = wave & 1;
  const int r0 = blockIdx.x * 128 + wy * 32;    // padded row base (2 m-tiles)
  const int c0 = blockIdx.y * 64 + wx * 32;     // col base (2 n-tiles)

  v8f acc00 = {}, acc01 = {}, acc10 = {}, acc11 = {};
  const bf16* a0row = A + (size_t)(r0 + lp) * DIM + 8 * half;
  const bf16* a1row = A + (size_t)(r0 + 16 + lp) * DIM + 8 * half;
  const bf16* w0row = W + (size_t)(c0 + lp) * DIM + 8 * half;
  const bf16* w1row = W + (size_t)(c0 + 16 + lp) * DIM + 8 * half;

#pragma unroll
  for (int k = 0; k < DIM; k += 32) {
    v16bf a0 = load_frag(a0row + k);
    v16bf a1 = load_frag(a1row + k);
    v16bf b0 = load_frag(w0row + k);
    v16bf b1 = load_frag(w1row + k);
    acc00 = WMMA_BF16(a0, b0, acc00);
    acc01 = WMMA_BF16(a0, b1, acc01);
    acc10 = WMMA_BF16(a1, b0, acc10);
    acc11 = WMMA_BF16(a1, b1, acc11);
  }

#pragma unroll
  for (int mt = 0; mt < 2; mt++) {
#pragma unroll
    for (int nt = 0; nt < 2; nt++) {
      v8f acc = mt ? (nt ? acc11 : acc10) : (nt ? acc01 : acc00);
      const int col = c0 + nt * 16 + lp;
      const float bia = bias[col];
#pragma unroll
      for (int r = 0; r < 8; r++) {
        const int row = r0 + mt * 16 + r + 8 * half;   // padded row
        float v = acc[r] + bia;
        if (RELU) v = v > 0.f ? v : 0.f;
        if (RES)  v += res[(size_t)row * DIM + col];
        if (OUTF) {
          if (TO_DOUT) {
            int bb = row / NP, n = row % NP;
            if (n < N_NODES)
              outf[((size_t)bb * N_NODES + n) * DIM + col] = v;
          } else {
            outf[(size_t)row * DIM + col] = v;
          }
        }
        if (OUTBF) outbf[(size_t)row * DIM + col] = f2bf(v);
        if (OUTXT) {
          int bb = row / NP, n = row % NP;
          xt[((size_t)bb * DIM + col) * NPX + n] = f2bf(v);
        }
      }
    }
  }
}

// ---------------------------------------------------------------------------
// Fused attention/aggregate:
//   out[i,:] = ( sum_j s_ij * x[j,:] ) / ( sum_j s_ij )
//   s_ij = (sigmoid(q_i . x_j) + 1e-5*[i==j]) * adjmask(i,j)
// Block: one (batch, 16-row i-tile). 8 waves: each computes one 16x16 S
// subtile per 128-wide j-chunk (WMMA bf16), masks branchlessly, writes bf16 S
// to LDS, reduces row sums via shfl_xor + ds_add_f32; then each wave
// aggregates a 32-wide D slice with WMMA over K=128: A from LDS
// (ds_load_b128 pairs), B from transposed XT (contiguous b128, L2-hot).
// ---------------------------------------------------------------------------
__global__ __launch_bounds__(256)
void attn_kernel(const bf16* __restrict__ Q, const bf16* __restrict__ X,
                 const bf16* __restrict__ XT, const float* __restrict__ adj,
                 bf16* __restrict__ out)
{
  __shared__ bf16  s_lds[16 * 128];   // S tile: [row i][k=j-in-chunk]
  __shared__ float denom[16];

  const int lane = threadIdx.x & 31;
  const int wave = threadIdx.x >> 5;
  const int lp = lane & 15, half = lane >> 4;
  const int b  = blockIdx.x / 63;
  const int i0 = (blockIdx.x % 63) * 16;
  const size_t xbase  = (size_t)b * NP * DIM;
  const size_t xtbase = (size_t)b * DIM * NPX;

  if (threadIdx.x < 16) denom[threadIdx.x] = 0.f;

  // Preload q A-fragments for rows i0..i0+15 (reused for every j-chunk).
  const bf16* qrow = Q + xbase + (size_t)(i0 + lp) * DIM + 8 * half;
  v16bf qf[8];
#pragma unroll
  for (int k = 0; k < 8; k++) qf[k] = load_frag(qrow + k * 32);

  v8f num0 = {}, num1 = {};
  const int c0 = wave * 32;           // D slice owned by this wave
  const bf16* xtr0 = XT + xtbase + (size_t)(c0 + lp) * NPX;
  const bf16* xtr1 = XT + xtbase + (size_t)(c0 + 16 + lp) * NPX;

  __syncthreads();

  for (int ch = 0; ch < 8; ch++) {
    const int j0 = ch * 128 + wave * 16;   // this wave's S subtile
    if (j0 < NP) {
      v8f sacc = {};
      const bf16* xrow = X + xbase + (size_t)(j0 + lp) * DIM + 8 * half;
#pragma unroll
      for (int k = 0; k < 8; k++) {
        v16bf xf = load_frag(xrow + k * 32);
        sacc = WMMA_BF16(qf[k], xf, sacc);
      }
      const int j  = j0 + lp;                              // this lane's column
      const int jc = j < N_NODES ? j : N_NODES - 1;        // clamped (branchless)
      const float* adjb = adj + (size_t)b * N_NODES * N_NODES + jc;
      // Unconditional clamped adjacency loads (coalesced across lanes).
      float av[8];
#pragma unroll
      for (int r = 0; r < 8; r++) {
        const int i  = i0 + r + 8 * half;
        const int ic = i < N_NODES ? i : N_NODES - 1;
        av[r] = adjb[(size_t)ic * N_NODES];
      }
#pragma unroll
      for (int r = 0; r < 8; r++) {
        const int i = i0 + r + 8 * half;
        const bool valid = (i < N_NODES) & (j < N_NODES);
        const bool diag  = (i == j);
        const float am  = diag ? 1.f : av[r];
        const float sig = 1.f / (1.f + __expf(-sacc[r]));
        const float v = valid ? (sig + (diag ? 1e-5f : 0.f)) * am : 0.f;
        sacc[r] = v;
        s_lds[(r + 8 * half) * 128 + wave * 16 + lp] = f2bf(v);
      }
      // Row-sum across the 16 lanes of this half-wave, add to denominator.
#pragma unroll
      for (int r = 0; r < 8; r++) {
        float v = sacc[r];
        v += __shfl_xor(v, 1, 16);
        v += __shfl_xor(v, 2, 16);
        v += __shfl_xor(v, 4, 16);
        v += __shfl_xor(v, 8, 16);
        if (lp == 0) atomicAdd(&denom[r + 8 * half], v);
      }
    } else {
      // tail chunk: zero-fill so aggregation WMMA reads zeros
#pragma unroll
      for (int r = 0; r < 8; r++)
        s_lds[(r + 8 * half) * 128 + wave * 16 + lp] = f2bf(0.f);
    }
    __syncthreads();

    // Aggregate: numer += S(16x128) @ X(128 x 32-slice), B from XT (contiguous)
#pragma unroll
    for (int kk = 0; kk < 4; kk++) {
      v16bf af = load_frag(&s_lds[lp * 128 + kk * 32 + 8 * half]);
      const int kbase = ch * 128 + kk * 32 + 8 * half;  // j base; max read 1023 < NPX
      v16bf b0 = load_frag(xtr0 + kbase);
      v16bf b1 = load_frag(xtr1 + kbase);
      num0 = WMMA_BF16(af, b0, num0);
      num1 = WMMA_BF16(af, b1, num1);
    }
    __syncthreads();
  }

  // Normalize and store (pad rows get denom==0 -> store 0, keeps buffers finite)
#pragma unroll
  for (int r = 0; r < 8; r++) {
    const float d = denom[r + 8 * half];
    const float inv = (d > 0.f) ? 1.f / d : 0.f;
    const int row = i0 + r + 8 * half;
    out[xbase + (size_t)row * DIM + c0 + lp]      = f2bf(num0[r] * inv);
    out[xbase + (size_t)row * DIM + c0 + 16 + lp] = f2bf(num1[r] * inv);
  }
}

// ---------------------------------------------------------------------------
__global__ void cvt_f32_bf16_kernel(const float* __restrict__ src,
                                    bf16* __restrict__ dst, int n) {
  int i = blockIdx.x * blockDim.x + threadIdx.x;
  if (i < n) dst[i] = f2bf(src[i]);
}

__global__ void prep_x_kernel(const float* __restrict__ xin,
                              float* __restrict__ xf, bf16* __restrict__ xb,
                              bf16* __restrict__ xt) {
  int idx = blockIdx.x * blockDim.x + threadIdx.x;   // over MP*DIM
  if (idx >= MP * DIM) return;
  int row = idx / DIM, d = idx % DIM;
  int b = row / NP, n = row % NP;
  float v = (n < N_NODES) ? xin[((size_t)b * N_NODES + n) * DIM + d] : 0.f;
  bf16 h = f2bf(v);
  xf[idx] = v;
  xb[idx] = h;
  xt[((size_t)b * DIM + d) * NPX + n] = h;
}

// ---------------------------------------------------------------------------
extern "C" void kernel_launch(void* const* d_in, const int* in_sizes, int n_in,
                              void* d_out, int out_size, void* d_ws, size_t ws_size,
                              hipStream_t stream) {
  (void)in_sizes; (void)n_in; (void)out_size; (void)ws_size;
  const float* x_in   = (const float*)d_in[0];
  const float* adj    = (const float*)d_in[1];
  const float* W_attn = (const float*)d_in[2];
  const float* b_attn = (const float*)d_in[3];
  const float* W0     = (const float*)d_in[4];
  const float* b0     = (const float*)d_in[5];
  const float* W1     = (const float*)d_in[6];
  const float* b1     = (const float*)d_in[7];
  const float* Wf     = (const float*)d_in[8];
  const float* bfin   = (const float*)d_in[9];

  char* ws = (char*)d_ws;
  bf16*  wbf    = (bf16*)ws;  ws += (size_t)(3 * NL + 1) * DD * 2;
  bf16*  x_bf   = (bf16*)ws;  ws += (size_t)MP * DIM * 2;
  bf16*  q_bf   = (bf16*)ws;  ws += (size_t)MP * DIM * 2;
  bf16*  agg_bf = (bf16*)ws;  ws += (size_t)MP * DIM * 2;
  bf16*  h_bf   = (bf16*)ws;  ws += (size_t)MP * DIM * 2;
  bf16*  xt_bf  = (bf16*)ws;  ws += (size_t)NB * DIM * NPX * 2;
  float* xA     = (float*)ws; ws += (size_t)MP * DIM * 4;
  float* xB     = (float*)ws;

  const int wtot = NL * DD;
  cvt_f32_bf16_kernel<<<(wtot + 255) / 256, 256, 0, stream>>>(W_attn, wbf, wtot);
  cvt_f32_bf16_kernel<<<(wtot + 255) / 256, 256, 0, stream>>>(W0, wbf + (size_t)NL * DD, wtot);
  cvt_f32_bf16_kernel<<<(wtot + 255) / 256, 256, 0, stream>>>(W1, wbf + (size_t)2 * NL * DD, wtot);
  cvt_f32_bf16_kernel<<<(DD + 255) / 256, 256, 0, stream>>>(Wf, wbf + (size_t)3 * NL * DD, DD);

  prep_x_kernel<<<(MP * DIM + 255) / 256, 256, 0, stream>>>(x_in, xA, x_bf, xt_bf);

  dim3 ggrid(MP / 128, DIM / 64);
  float* cur = xA; float* nxt = xB;
  for (int l = 0; l < NL; l++) {
    // q = x @ W_attn^T + b  (bf16 out)
    gemm_kernel<0,1,0,0,0,0><<<ggrid, 256, 0, stream>>>(
        x_bf, wbf + (size_t)l * DD, b_attn + l * DIM, nullptr,
        nullptr, q_bf, nullptr);
    // fused masked-sigmoid attention + normalized aggregate
    attn_kernel<<<NB * 63, 256, 0, stream>>>(q_bf, x_bf, xt_bf, adj, agg_bf);
    // h = relu(agg @ W0^T + b0)  (bf16 out)
    gemm_kernel<0,1,0,1,0,0><<<ggrid, 256, 0, stream>>>(
        agg_bf, wbf + (size_t)(NL + l) * DD, b0 + l * DIM, nullptr,
        nullptr, h_bf, nullptr);
    // x = relu(h @ W1^T + b1) + x0  (f32 + bf16 row-major + bf16 transposed)
    gemm_kernel<1,1,1,1,1,0><<<ggrid, 256, 0, stream>>>(
        h_bf, wbf + (size_t)(2 * NL + l) * DD, b1 + l * DIM, cur,
        nxt, x_bf, xt_bf);
    float* t = cur; cur = nxt; nxt = t;
  }
  // final: out = x @ Wf^T + bf  (store only real rows, unpadded)
  gemm_kernel<1,0,0,0,0,1><<<ggrid, 256, 0, stream>>>(
      x_bf, wbf + (size_t)3 * NL * DD, bfin, nullptr,
      (float*)d_out, nullptr, nullptr);
}
